// Attention_84473416778449
// MI455X (gfx1250) — matrix operand order repaired and checked by reading, not compiled, not run on hardware
//
#include <hip/hip_runtime.h>

typedef __bf16 bf16;
typedef __bf16 v16bf __attribute__((ext_vector_type(16)));
typedef __bf16 v8bf  __attribute__((ext_vector_type(8)));
typedef float  v8f   __attribute__((ext_vector_type(8)));
typedef unsigned int v4u __attribute__((ext_vector_type(4)));
typedef unsigned int v8u __attribute__((ext_vector_type(8)));

#define NB 2
#define NS 2048
#define ND 1024
#define NH 16
#define HD 64

#define LDA 40   // 32 + 8 pad (bf16 elems), 16B-aligned stride
#define LDK 72   // 64 + 8 pad (bf16 elems) == TDM pad: 4 DWORDs per 32 DWORDs

// ---------------------------------------------------------------------------
// WMMA operand loaders (layouts per CDNA5 ISA 7.12.2, wave32)
// ---------------------------------------------------------------------------
__device__ __forceinline__ v16bf load_A(const bf16* base, int ld) {
    int lane = threadIdx.x & 31;
    int row  = lane & 15;
    int k0   = (lane >> 4) * 8;
    const bf16* p = base + row * ld + k0;
    v8bf lo = *(const v8bf*)(p);
    v8bf hi = *(const v8bf*)(p + 16);
    return __builtin_shufflevector(lo, hi, 0,1,2,3,4,5,6,7,8,9,10,11,12,13,14,15);
}

__device__ __forceinline__ v16bf load_B(const bf16* base, int ld) {
    int lane = threadIdx.x & 31;
    int col  = lane & 15;
    int koff = (lane >> 4) * 8;
    const bf16* p = base + col * ld + koff;
    v8bf lo = *(const v8bf*)(p);
    v8bf hi = *(const v8bf*)(p + 16);
    return __builtin_shufflevector(lo, hi, 0,1,2,3,4,5,6,7,8,9,10,11,12,13,14,15);
}

__device__ __forceinline__ v8f wmma_bf16(v16bf a, v16bf b, v8f c) {
    return __builtin_amdgcn_wmma_f32_16x16x32_bf16(false, a, false, b,
                                                   (short)0, c, false, false);
}

// ---------------------------------------------------------------------------
// Tensor Data Mover: 2D bf16 tile global -> LDS with hardware row padding
// (4 DWORDs after every 32 DWORDs => 72-element LDS rows == LDK).
// ---------------------------------------------------------------------------
__device__ __forceinline__ void tdm_load_2d_bf16(
    unsigned lds_addr, const void* gaddr,
    unsigned tile_d0, unsigned tile_d1,
    unsigned long long stride0_elems)
{
    unsigned long long ga = (unsigned long long)(size_t)gaddr;
    v4u g0;
    g0[0] = 1u;                                         // count = 1 valid D#
    g0[1] = lds_addr;                                   // LDS byte address
    g0[2] = (unsigned)(ga & 0xffffffffu);               // global_addr[31:0]
    g0[3] = (unsigned)((ga >> 32) & 0x01ffffffu)        // global_addr[56:32]
          | (2u << 30);                                 // type = 2 (image)
    v8u g1;
    g1[0] = (1u << 16)                                  // data_size = 1 (2B)
          | (1u << 20)                                  // pad_enable
          | (4u << 22)                                  // pad_interval: 32 DW
          | (3u << 25);                                 // pad_amount: 4 DW
    g1[1] = (tile_d0 & 0xffffu) << 16;                  // tensor_dim0[15:0]
    g1[2] = (tile_d0 >> 16)                             // tensor_dim0[31:16]
          | ((tile_d1 & 0xffffu) << 16);                // tensor_dim1[15:0]
    g1[3] = (tile_d1 >> 16)
          | (tile_d0 << 16);                            // tile_dim0
    g1[4] = tile_d1 & 0xffffu;                          // tile_dim1 (dim2 = 0)
    g1[5] = (unsigned)(stride0_elems & 0xffffffffu);    // dim0 stride
    g1[6] = (unsigned)((stride0_elems >> 32) & 0xffffu);
    g1[7] = 0u;
    asm volatile("tensor_load_to_lds %0, %1" :: "s"(g0), "s"(g1) : "memory");
}

// fp32 -> bf16 convert + LDS store of 16 staged elements
__device__ __forceinline__ void cvt_store16(const float* v, bf16* dst) {
    v8bf lo, hi;
    #pragma unroll
    for (int i = 0; i < 8; i++) { lo[i] = (bf16)v[i]; hi[i] = (bf16)v[8 + i]; }
    *(v8bf*)dst       = lo;
    *(v8bf*)(dst + 8) = hi;
}

// ---------------------------------------------------------------------------
// Kernel 1: q/k/v = x @ w{q,k,v}^T  (blockIdx.z picks weight)
// 128x128 WG tile, 8 waves = 4Mx2N, 2x4 WMMA each; double-buffered LDS,
// one barrier per K step; global loads for step k+1 overlap WMMAs of step k.
// ---------------------------------------------------------------------------
__global__ __launch_bounds__(256) void qkv_gemm_kernel(
    const float* __restrict__ x,
    const float* __restrict__ wq, const float* __restrict__ wk,
    const float* __restrict__ wv,
    bf16* __restrict__ qbuf, bf16* __restrict__ kbuf, bf16* __restrict__ vT)
{
    __shared__ bf16 As[2][128 * LDA];
    __shared__ bf16 Ws[2][128 * LDA];

    const int mBase = blockIdx.x * 128;
    const int nBase = blockIdx.y * 128;
    const int z     = blockIdx.z;
    const float* w  = (z == 0) ? wq : ((z == 1) ? wk : wv);

    const int t    = threadIdx.x;
    const int wave = t >> 5;
    const int lane = t & 31;
    const int wm   = wave >> 1;   // 0..3 -> M offset wm*32
    const int wn   = wave & 1;    // 0..1 -> N offset wn*64
    const int row  = t >> 1, half = t & 1;

    const float* srcX = x + (size_t)(mBase + row) * ND + half * 16;
    const float* srcW = w + (size_t)(nBase + row) * ND + half * 16;
    bf16* dstA = &As[0][0] + row * LDA + half * 16;   // [cur] selected by +stride
    bf16* dstW = &Ws[0][0] + row * LDA + half * 16;
    const int bufStride = 128 * LDA;

    v8f acc[2][4] = {};

    // prologue: stage K-step 0 into buffer 0
    {
        float xa[16], wa[16];
        #pragma unroll
        for (int i = 0; i < 16; i++) { xa[i] = srcX[i]; wa[i] = srcW[i]; }
        cvt_store16(xa, dstA);
        cvt_store16(wa, dstW);
    }
    __syncthreads();

    for (int kt = 0; kt < ND; kt += 32) {
        const int cur = (kt >> 5) & 1;
        const bool hasNext = (kt + 32) < ND;

        // issue global loads for next K step (overlap with WMMAs below)
        float xa[16], wa[16];
        if (hasNext) {
            const float* nx = srcX + kt + 32;
            const float* nw = srcW + kt + 32;
            __builtin_prefetch(nx + 32, 0, 3);
            __builtin_prefetch(nw + 32, 0, 3);
            #pragma unroll
            for (int i = 0; i < 16; i++) { xa[i] = nx[i]; wa[i] = nw[i]; }
        }

        // compute from buffer[cur]
        const bf16* curA = &As[cur][0];
        const bf16* curW = &Ws[cur][0];
        v16bf bt[4];
        #pragma unroll
        for (int ni = 0; ni < 4; ni++)
            bt[ni] = load_B(curW + (wn * 64 + ni * 16) * LDA, LDA);
        #pragma unroll
        for (int mi = 0; mi < 2; mi++) {
            v16bf a = load_A(curA + (wm * 32 + mi * 16) * LDA, LDA);
            #pragma unroll
            for (int ni = 0; ni < 4; ni++)
                acc[mi][ni] = wmma_bf16(a, bt[ni], acc[mi][ni]);
        }

        // store staged data into buffer[cur^1]
        if (hasNext) {
            cvt_store16(xa, dstA + (cur ^ 1) * bufStride);
            cvt_store16(wa, dstW + (cur ^ 1) * bufStride);
        }
        __syncthreads();
    }

    // write-out; z-branch hoisted (uniform) so store loops stay clean
    const int khalf = lane >> 4, col = lane & 15;
    if (z == 2) {
        #pragma unroll
        for (int mi = 0; mi < 2; mi++)
            #pragma unroll
            for (int ni = 0; ni < 4; ni++)
                #pragma unroll
                for (int r = 0; r < 8; r++) {
                    int m = mBase + wm * 32 + mi * 16 + r + (khalf << 3);
                    int n = nBase + wn * 64 + ni * 16 + col;
                    int b = m >> 11, s = m & (NS - 1);
                    int h = n >> 6,  d = n & (HD - 1);
                    vT[((size_t)((b << 4) + h) * HD + d) * NS + s] =
                        (bf16)acc[mi][ni][r];
                }
    } else {
        bf16* dst = (z == 0) ? qbuf : kbuf;
        #pragma unroll
        for (int mi = 0; mi < 2; mi++)
            #pragma unroll
            for (int ni = 0; ni < 4; ni++)
                #pragma unroll
                for (int r = 0; r < 8; r++) {
                    int m = mBase + wm * 32 + mi * 16 + r + (khalf << 3);
                    int n = nBase + wn * 64 + ni * 16 + col;
                    int b = m >> 11, s = m & (NS - 1);
                    int h = n >> 6,  d = n & (HD - 1);
                    dst[((size_t)((b << 4) + h) * NS + s) * HD + d] =
                        (bf16)acc[mi][ni][r];
                }
    }
}

// ---------------------------------------------------------------------------
// Kernel 2: RoPE in place on q,k ([B,H,S,HD] bf16), pairs (d, d+32)
// ---------------------------------------------------------------------------
__global__ __launch_bounds__(256) void rope_kernel(
    bf16* __restrict__ qbuf, bf16* __restrict__ kbuf,
    const float* __restrict__ cosb, const float* __restrict__ sinb)
{
    int idx = blockIdx.x * 256 + threadIdx.x;   // B*H*S*32 pairs
    int d  = idx & 31;
    int s  = (idx >> 5) & (NS - 1);
    int bh = idx >> 16;
    size_t base = ((size_t)bh * NS + s) * HD;
    float c1 = cosb[s * HD + d],      s1 = sinb[s * HD + d];
    float c2 = cosb[s * HD + d + 32], s2 = sinb[s * HD + d + 32];

    float q1 = (float)qbuf[base + d], q2 = (float)qbuf[base + d + 32];
    qbuf[base + d]      = (bf16)(q1 * c1 - q2 * s1);
    qbuf[base + d + 32] = (bf16)(q2 * c2 + q1 * s2);

    float k1 = (float)kbuf[base + d], k2 = (float)kbuf[base + d + 32];
    kbuf[base + d]      = (bf16)(k1 * c1 - k2 * s1);
    kbuf[base + d + 32] = (bf16)(k2 * c2 + k1 * s2);
}

// ---------------------------------------------------------------------------
// Kernel 3: causal flash attention per (b,h). WG = 128 q rows (8 waves x 16).
// K/V 64-row blocks double-buffered via the Tensor Data Mover: wave 0 issues
// the DMA for block kb+1 into the alternate buffers BEFORE waiting
// s_wait_tensorcnt(2) for block kb (TDM ops complete in order), so the DMA
// overlaps score/softmax/PV WMMA work. Probs tile is wave-private (LDS ops
// are in-order per wave) -> only 2 barriers per iteration.
// ---------------------------------------------------------------------------
__global__ __launch_bounds__(256) void attn_kernel(
    const bf16* __restrict__ qbuf, const bf16* __restrict__ kbuf,
    const bf16* __restrict__ vT, bf16* __restrict__ attn)
{
    __shared__ bf16 Kt[2][64 * LDK];     // [k_idx][d]  -> scores B operand
    __shared__ bf16 Vt[2][64 * LDK];     // [d][k_idx]  -> PV B operand
    __shared__ bf16 Pt[8 * 16 * LDK];    // per-wave probs tile [16][64]

    const int qblk = blockIdx.x;         // 0..15  (128 q rows each)
    const int bh   = blockIdx.y;         // 0..31
    const int t    = threadIdx.x;
    const int wave = t >> 5;
    const int lane = t & 31;
    const int khalf = lane >> 4, col = lane & 15;
    const int qbase = qblk * 128 + wave * 16;

    // q rows for this wave, directly from global as A operands (d 0..31, 32..63)
    const bf16* qrow = qbuf + ((size_t)bh * NS + qbase + col) * HD;
    const int k0 = khalf * 8;
    v16bf qA0, qA1;
    {
        v8bf lo = *(const v8bf*)(qrow + k0);
        v8bf hi = *(const v8bf*)(qrow + k0 + 16);
        qA0 = __builtin_shufflevector(lo, hi, 0,1,2,3,4,5,6,7,8,9,10,11,12,13,14,15);
        v8bf lo2 = *(const v8bf*)(qrow + 32 + k0);
        v8bf hi2 = *(const v8bf*)(qrow + 32 + k0 + 16);
        qA1 = __builtin_shufflevector(lo2, hi2, 0,1,2,3,4,5,6,7,8,9,10,11,12,13,14,15);
    }

    v8f oacc[4] = {};
    float m_r[8], l_r[8];
    #pragma unroll
    for (int r = 0; r < 8; r++) { m_r[r] = -1e30f; l_r[r] = 0.f; }

    const bf16* gK = kbuf + (size_t)bh * NS * HD;   // rows [k][d]
    const bf16* gV = vT + (size_t)bh * HD * NS;     // rows [d][s]
    const unsigned ldsK[2] = { (unsigned)(size_t)(void*)&Kt[0][0],
                               (unsigned)(size_t)(void*)&Kt[1][0] };
    const unsigned ldsV[2] = { (unsigned)(size_t)(void*)&Vt[0][0],
                               (unsigned)(size_t)(void*)&Vt[1][0] };

    const int nkb = 2 * qblk + 2;        // 64-wide k blocks covering causal span
    if (wave == 0) {
        tdm_load_2d_bf16(ldsK[0], gK, 64, 64, HD);
        tdm_load_2d_bf16(ldsV[0], gV, 64, 64, NS);
    }

    for (int kb = 0; kb < nkb; kb++) {
        const int kstart = kb * 64;
        const int cur = kb & 1;
        if (wave == 0) {
            if (kb + 1 < nkb) {
                // prefetch next block into alternate buffers, then wait for
                // the current block's pair (in-order completion => cnt<=2).
                tdm_load_2d_bf16(ldsK[cur ^ 1], gK + (size_t)(kstart + 64) * HD,
                                 64, 64, HD);
                tdm_load_2d_bf16(ldsV[cur ^ 1], gV + (kstart + 64),
                                 64, 64, NS);
                __builtin_amdgcn_s_wait_tensorcnt(2);
            } else {
                __builtin_amdgcn_s_wait_tensorcnt(0);
            }
        }
        __syncthreads();

        // scores: 16 q rows x 64 k cols, K-dim = HD = 64 (two chained WMMAs)
        v8f sc[4];
        #pragma unroll
        for (int n = 0; n < 4; n++) {
            v16bf b0 = load_B(&Kt[cur][(n * 16) * LDK], LDK);
            v16bf b1 = load_B(&Kt[cur][(n * 16) * LDK + 32], LDK);
            v8f s = {};
            s = wmma_bf16(qA0, b0, s);
            s = wmma_bf16(qA1, b1, s);
            sc[n] = s;
        }

        // scale + causal mask + online softmax (row lives across 16 lanes)
        #pragma unroll
        for (int r = 0; r < 8; r++) {
            int qg = qbase + r + (khalf << 3);
            float rmax = -1e30f;
            #pragma unroll
            for (int n = 0; n < 4; n++) {
                int kg = kstart + n * 16 + col;
                float v = sc[n][r] * 0.125f + ((kg <= qg) ? 0.f : -1e30f);
                sc[n][r] = v;
                rmax = fmaxf(rmax, v);
            }
            rmax = fmaxf(rmax, __shfl_xor(rmax, 1, 16));
            rmax = fmaxf(rmax, __shfl_xor(rmax, 2, 16));
            rmax = fmaxf(rmax, __shfl_xor(rmax, 4, 16));
            rmax = fmaxf(rmax, __shfl_xor(rmax, 8, 16));
            float mnew = fmaxf(m_r[r], rmax);
            float fr = __expf(m_r[r] - mnew);
            float rsum = 0.f;
            #pragma unroll
            for (int n = 0; n < 4; n++) {
                float p = __expf(sc[n][r] - mnew);
                sc[n][r] = p;
                rsum += p;
            }
            rsum += __shfl_xor(rsum, 1, 16);
            rsum += __shfl_xor(rsum, 2, 16);
            rsum += __shfl_xor(rsum, 4, 16);
            rsum += __shfl_xor(rsum, 8, 16);
            l_r[r] = l_r[r] * fr + rsum;
            m_r[r] = mnew;
            #pragma unroll
            for (int nd = 0; nd < 4; nd++) oacc[nd][r] *= fr;
        }

        // probs -> per-wave LDS tile (wave-private; LDS in-order per wave)
        bf16* pw = &Pt[wave * 16 * LDK];
        #pragma unroll
        for (int r = 0; r < 8; r++) {
            int rr = r + (khalf << 3);
            #pragma unroll
            for (int n = 0; n < 4; n++)
                pw[rr * LDK + n * 16 + col] = (bf16)sc[n][r];
        }

        // PV: out[16 x 64] += probs[16 x 64] @ V[64 x 64]
        #pragma unroll
        for (int kc = 0; kc < 2; kc++) {
            v16bf a = load_A(pw + kc * 32, LDK);
            #pragma unroll
            for (int nd = 0; nd < 4; nd++) {
                v16bf b = load_B(&Vt[cur][(nd * 16) * LDK + kc * 32], LDK);
                oacc[nd] = wmma_bf16(a, b, oacc[nd]);
            }
        }
        __syncthreads();   // buffers[cur] free for TDM two iterations later
    }

    // epilogue: divide by l, write attn [B,S,D]
    const int b = bh >> 4, h = bh & 15;
    #pragma unroll
    for (int nd = 0; nd < 4; nd++) {
        #pragma unroll
        for (int r = 0; r < 8; r++) {
            int s = qbase + r + (khalf << 3);
            int d = nd * 16 + col;
            attn[((size_t)b * NS + s) * ND + h * HD + d] =
                (bf16)(oacc[nd][r] / l_r[r]);
        }
    }
}

// ---------------------------------------------------------------------------
// Kernel 4: out = attn(bf16) @ wo^T -> fp32 d_out (128x128 WG tile,
// double-buffered like kernel 1)
// ---------------------------------------------------------------------------
__global__ __launch_bounds__(256) void out_gemm_kernel(
    const bf16* __restrict__ attn, const float* __restrict__ wo,
    float* __restrict__ out)
{
    __shared__ bf16 As[2][128 * LDA];
    __shared__ bf16 Ws[2][128 * LDA];

    const int mBase = blockIdx.x * 128;
    const int nBase = blockIdx.y * 128;
    const int t    = threadIdx.x;
    const int wave = t >> 5;
    const int lane = t & 31;
    const int wm   = wave >> 1;
    const int wn   = wave & 1;
    const int row  = t >> 1, half = t & 1;

    const bf16*  srcA = attn + (size_t)(mBase + row) * ND + half * 16;
    const float* srcW = wo   + (size_t)(nBase + row) * ND + half * 16;
    bf16* dstA = &As[0][0] + row * LDA + half * 16;
    bf16* dstW = &Ws[0][0] + row * LDA + half * 16;
    const int bufStride = 128 * LDA;

    v8f acc[2][4] = {};

    {
        v8bf a0 = *(const v8bf*)(srcA);
        v8bf a1 = *(const v8bf*)(srcA + 8);
        float wa[16];
        #pragma unroll
        for (int i = 0; i < 16; i++) wa[i] = srcW[i];
        *(v8bf*)dstA       = a0;
        *(v8bf*)(dstA + 8) = a1;
        cvt_store16(wa, dstW);
    }
    __syncthreads();

    for (int kt = 0; kt < ND; kt += 32) {
        const int cur = (kt >> 5) & 1;
        const bool hasNext = (kt + 32) < ND;

        v8bf a0, a1;
        float wa[16];
        if (hasNext) {
            const bf16* nA = srcA + kt + 32;
            const float* nW = srcW + kt + 32;
            __builtin_prefetch(nW + 32, 0, 3);
            a0 = *(const v8bf*)(nA);
            a1 = *(const v8bf*)(nA + 8);
            #pragma unroll
            for (int i = 0; i < 16; i++) wa[i] = nW[i];
        }

        const bf16* curA = &As[cur][0];
        const bf16* curW = &Ws[cur][0];
        v16bf bt[4];
        #pragma unroll
        for (int ni = 0; ni < 4; ni++)
            bt[ni] = load_B(curW + (wn * 64 + ni * 16) * LDA, LDA);
        #pragma unroll
        for (int mi = 0; mi < 2; mi++) {
            v16bf a = load_A(curA + (wm * 32 + mi * 16) * LDA, LDA);
            #pragma unroll
            for (int ni = 0; ni < 4; ni++)
                acc[mi][ni] = wmma_bf16(a, bt[ni], acc[mi][ni]);
        }

        if (hasNext) {
            bf16* dA = dstA + (cur ^ 1) * bufStride;
            *(v8bf*)dA       = a0;
            *(v8bf*)(dA + 8) = a1;
            cvt_store16(wa, dstW + (cur ^ 1) * bufStride);
        }
        __syncthreads();
    }

    const int khalf = lane >> 4, col = lane & 15;
    #pragma unroll
    for (int mi = 0; mi < 2; mi++)
        #pragma unroll
        for (int ni = 0; ni < 4; ni++)
            #pragma unroll
            for (int r = 0; r < 8; r++) {
                int m = mBase + wm * 32 + mi * 16 + r + (khalf << 3);
                int n = nBase + wn * 64 + ni * 16 + col;
                out[(size_t)m * ND + n] = acc[mi][ni][r];
            }
}

// ---------------------------------------------------------------------------
extern "C" void kernel_launch(void* const* d_in, const int* in_sizes, int n_in,
                              void* d_out, int out_size, void* d_ws, size_t ws_size,
                              hipStream_t stream)
{
    const float* x    = (const float*)d_in[0];
    const float* cosb = (const float*)d_in[1];
    const float* sinb = (const float*)d_in[2];
    // d_in[3] = mask (unused; causal mask applied analytically)
    const float* wq   = (const float*)d_in[4];
    const float* wk   = (const float*)d_in[5];
    const float* wv   = (const float*)d_in[6];
    const float* wo   = (const float*)d_in[7];
    float* out = (float*)d_out;

    char* ws = (char*)d_ws;
    const size_t qkvBytes = (size_t)NB * NH * NS * HD * sizeof(bf16);   // 8 MiB each
    bf16* qbuf = (bf16*)(ws);
    bf16* kbuf = (bf16*)(ws + qkvBytes);
    bf16* vT   = (bf16*)(ws + 2 * qkvBytes);
    bf16* attn = (bf16*)(ws + 3 * qkvBytes);                            // 8 MiB

    qkv_gemm_kernel<<<dim3(NB * NS / 128, ND / 128, 3), 256, 0, stream>>>(
        x, wq, wk, wv, qbuf, kbuf, vT);

    rope_kernel<<<(NB * NH * NS * 32) / 256, 256, 0, stream>>>(
        qbuf, kbuf, cosb, sinb);

    attn_kernel<<<dim3(NS / 128, NB * NH), 256, 0, stream>>>(
        qbuf, kbuf, vT, attn);

    out_gemm_kernel<<<dim3(NB * NS / 128, ND / 128), 256, 0, stream>>>(
        attn, wo, out);
}
